// TemporalGAT_RNN_68118181315174
// MI455X (gfx1250) — compile-verified, matrix operand order checked by reading, not run on hardware
//
#include <hip/hip_runtime.h>
#include <hip/hip_bf16.h>

// ---------------------------------------------------------------------------
// CDNA5 (gfx1250) WMMA plumbing: wave32, V_WMMA_F32_16X16X32_BF16
// Fragment layouts per ISA 7.12.2 (16-bit A 16x32 / B 32x16 / f32 C 16x16).
// All GEMM operands are kept "fragment-resident": each lane's 16 bf16 elements
// are contiguous 32 bytes, so fragment loads are 2x b128 instead of 16x u16.
// ---------------------------------------------------------------------------
typedef __attribute__((ext_vector_type(16))) __bf16         v16bf;
typedef __attribute__((ext_vector_type(16))) unsigned short v16u;
typedef __attribute__((ext_vector_type(8)))  float          v8f;

static __device__ inline unsigned short f2bf(float x) {
  unsigned u = __float_as_uint(x);
  unsigned r = u + 0x7fffu + ((u >> 16) & 1u);   // round-to-nearest-even
  return (unsigned short)(r >> 16);
}
static __device__ inline float bf2f(unsigned v) {
  return __uint_as_float(v << 16);
}
// fast sigmoid: single v_rcp_f32 instead of IEEE div sequence
static __device__ inline float sigf(float x) {
  return __builtin_amdgcn_rcpf(1.0f + __expf(-x));
}
// fast tanh: 1 - 2/(e^{2x}+1); saturates correctly to +/-1 at large |x|
static __device__ inline float tanh_fast(float x) {
  return 1.0f - 2.0f * __builtin_amdgcn_rcpf(__expf(2.0f * x) + 1.0f);
}

static __device__ inline v8f wmma_bf16f32(v16u a, v16u b, v8f c) {
  // (neg_a, A, neg_b, B, c_mod, C, reuse_a, reuse_b)
  return __builtin_amdgcn_wmma_f32_16x16x32_bf16(
      false, __builtin_bit_cast(v16bf, a),
      false, __builtin_bit_cast(v16bf, b),
      (short)0, c, false, false);
}

// Inverse of the WMMA 16-bit operand mapping: element (m,k) of a 16xK tile
// -> flat index in fragment-swizzled storage [(kblk*32 + lane)*16 + e].
static __device__ inline int afrag_idx(int m, int k) {
  int kblk = k >> 5, kin = k & 31;
  int lane = m + (((kin >> 3) & 1) << 4);
  int e = (kin & 7) | ((kin >> 4) << 3);
  return (kblk * 32 + lane) * 16 + e;
}

// A fragment from LDS tile stored in fragment order: one 32B load per lane.
static __device__ inline v16u load_afrag_swz_lds(const unsigned short* A,
                                                 int lane, int kblk) {
  return *(const v16u*)(A + (size_t)(kblk * 32 + lane) * 16);
}

// B fragment from a pre-swizzled global matrix: one 32B load per lane.
// Layout: B_swz[((kb*CT + ct)*32 + lane)*16 + e]
static __device__ inline v16u load_bfrag_swz(const unsigned short* __restrict__ B,
                                             int lane, int kb, int ct, int CT) {
  return *(const v16u*)(B + ((size_t)(kb * CT + ct) * 32 + lane) * 16);
}

// ---------------------------------------------------------------------------
// Small utility kernels
// ---------------------------------------------------------------------------
__global__ void k_zero(float* __restrict__ p, int n) {
  int i = blockIdx.x * blockDim.x + threadIdx.x;
  if (i < n) p[i] = 0.0f;
}

__global__ void k_cvt_bf16(const float* __restrict__ in,
                           unsigned short* __restrict__ out, int n) {
  int i = blockIdx.x * blockDim.x + threadIdx.x;
  if (i < n) out[i] = f2bf(in[i]);
}

// Swizzle an f32 matrix into WMMA-B fragment order (bf16).
// Logical B is [K][ncols]; in_n_major=1 means the source is [ncols][K].
__global__ void k_swizzleB(const float* __restrict__ in,
                           unsigned short* __restrict__ out,
                           int K, int ncols, int in_n_major) {
  int idx = blockIdx.x * blockDim.x + threadIdx.x;
  if (idx >= K * ncols) return;
  int e    = idx & 15;
  int lane = (idx >> 4) & 31;
  int rest = idx >> 9;               // kb*CT + ct
  int CT = ncols >> 4;
  int ct = rest % CT, kb = rest / CT;
  int n = ct * 16 + (lane & 15);
  int kbh = (lane < 16) ? 0 : 8;
  int k = kb * 32 + kbh + ((e < 8) ? e : (e + 8));
  float v = in_n_major ? in[(size_t)n * K + k] : in[(size_t)k * ncols + n];
  out[idx] = f2bf(v);
}

__global__ void k_softmax5(const float* __restrict__ a0,
                           const float* __restrict__ a1,
                           float* __restrict__ probs) {
  if (threadIdx.x == 0 && blockIdx.x == 0) {
    for (int s = 0; s < 2; ++s) {
      const float* a = s ? a1 : a0;
      float mx = a[0];
      for (int i = 1; i < 5; ++i) mx = fmaxf(mx, a[i]);
      float e[5], sum = 0.0f;
      for (int i = 0; i < 5; ++i) { e[i] = __expf(a[i] - mx); sum += e[i]; }
      float rs = __builtin_amdgcn_rcpf(sum);
      for (int i = 0; i < 5; ++i) probs[s * 8 + i] = e[i] * rs;
    }
  }
}

// ---------------------------------------------------------------------------
// GCN normalization
// ---------------------------------------------------------------------------
__global__ void k_deg_edges(const int* __restrict__ col,
                            const float* __restrict__ w,
                            float* __restrict__ deg, int E) {
  int e = blockIdx.x * blockDim.x + threadIdx.x;
  if (e < E) atomicAdd(&deg[col[e]], w[e]);
}

__global__ void k_dis(const float* __restrict__ deg, float* __restrict__ dis,
                      float* __restrict__ selfnorm, int N) {
  int n = blockIdx.x * blockDim.x + threadIdx.x;
  if (n < N) {
    float d = deg[n] + 1.0f;            // self-loop weight 1
    d = (d > 0.0f) ? d : 1.0f;
    float r = rsqrtf(d);
    dis[n] = r;
    selfnorm[n] = r * r;
  }
}

__global__ void k_edgenorm(const int* __restrict__ row, const int* __restrict__ col,
                           const float* __restrict__ w, const float* __restrict__ dis,
                           float* __restrict__ normw, int E) {
  int e = blockIdx.x * blockDim.x + threadIdx.x;
  if (e < E) normw[e] = dis[row[e]] * w[e] * dis[col[e]];
}

// ---------------------------------------------------------------------------
// xw = x[:,:,t] @ W{z,r,h}   (N x 16) x (16 x 64), three weights at once
// ---------------------------------------------------------------------------
__global__ void k_xw(const float* __restrict__ x, int t,
                     const float* __restrict__ Wz, const float* __restrict__ Wr,
                     const float* __restrict__ Wh,
                     float* __restrict__ xwz, float* __restrict__ xwr,
                     float* __restrict__ xwh, int N) {
  int idx = blockIdx.x * blockDim.x + threadIdx.x;
  if (idx >= N * 64) return;
  int n = idx >> 6, d = idx & 63;
  float az = 0.f, ar = 0.f, ah = 0.f;
#pragma unroll
  for (int i = 0; i < 16; ++i) {
    float xv = x[(n * 16 + i) * 5 + t];
    az += xv * Wz[i * 64 + d];
    ar += xv * Wr[i * 64 + d];
    ah += xv * Wh[i * 64 + d];
  }
  xwz[idx] = az; xwr[idx] = ar; xwh[idx] = ah;
}

// ---------------------------------------------------------------------------
// segment-sum scatter over edges + self loops (float4 granularity, 3 gates)
// ---------------------------------------------------------------------------
__global__ void k_scatter(const int* __restrict__ row, const int* __restrict__ col,
                          const float* __restrict__ normw,
                          const float* __restrict__ selfnorm,
                          const float* __restrict__ xwz, const float* __restrict__ xwr,
                          const float* __restrict__ xwh,
                          float* __restrict__ gz, float* __restrict__ gr,
                          float* __restrict__ gh, int E, int N) {
  int idx = blockIdx.x * blockDim.x + threadIdx.x;
  int total = (E + N) * 16;
  if (idx >= total) return;
  int d4 = (idx & 15) * 4;
  int e  = idx >> 4;
  int r, c; float w;
  if (e < E) { r = row[e]; c = col[e]; w = normw[e]; }
  else       { r = e - E;  c = r;      w = selfnorm[r]; }
  int rb = r * 64 + d4, cb = c * 64 + d4;
#pragma unroll
  for (int q = 0; q < 4; ++q) {
    atomicAdd(&gz[cb + q], w * xwz[rb + q]);
    atomicAdd(&gr[cb + q], w * xwr[rb + q]);
    atomicAdd(&gh[cb + q], w * xwh[rb + q]);
  }
}

// ---------------------------------------------------------------------------
// GRU-style gate block for a 16-node tile. 4 waves, each owns a 16x16 C tile.
// Z = sig([gcnz+bz | H0] Lz + lbz); R likewise; Ht = tanh([gcnh+bh | H0*R] Lh + lbh)
// Hacc += prob * (Z*H0 + (1-Z)*Ht)
// ---------------------------------------------------------------------------
__global__ __launch_bounds__(128) void k_gate(
    const float* __restrict__ gcnz, const float* __restrict__ gcnr,
    const float* __restrict__ gcnh,
    const float* __restrict__ bz, const float* __restrict__ br,
    const float* __restrict__ bh,
    const float* __restrict__ H0,
    const unsigned short* __restrict__ Lz, const unsigned short* __restrict__ Lr,
    const unsigned short* __restrict__ Lh,    // swizzled [K=128][64]
    const float* __restrict__ lbz, const float* __restrict__ lbr,
    const float* __restrict__ lbh,
    const float* __restrict__ probs, int t,
    float* __restrict__ Hacc) {
  __shared__ __align__(32) unsigned short As[16 * 128];
  __shared__ float Zs[16 * 64];
  __shared__ float Rs[16 * 64];
  __shared__ float H0s[16 * 64];

  int tid = threadIdx.x, lane = tid & 31, w = tid >> 5;
  int n0 = blockIdx.x * 16;
  float prob = probs[t];

  for (int idx = tid; idx < 16 * 64; idx += 128)
    H0s[idx] = H0[n0 * 64 + idx];
  for (int idx = tid; idx < 16 * 128; idx += 128) {
    int m = idx >> 7, k = idx & 127;
    float v = (k < 64) ? (gcnz[(n0 + m) * 64 + k] + bz[k])
                       : H0[(n0 + m) * 64 + (k - 64)];
    As[afrag_idx(m, k)] = f2bf(v);
  }
  __syncthreads();

  int col = w * 16 + (lane & 15);
  int rb  = (lane < 16) ? 0 : 8;

  // ---- Z ----
  v8f c = {};
#pragma unroll
  for (int kb = 0; kb < 4; ++kb) {
    v16u a = load_afrag_swz_lds(As, lane, kb);
    v16u b = load_bfrag_swz(Lz, lane, kb, w, 4);
    c = wmma_bf16f32(a, b, c);
  }
  {
    float lb = lbz[col];
#pragma unroll
    for (int v = 0; v < 8; ++v) Zs[(rb + v) * 64 + col] = sigf(c[v] + lb);
  }
  __syncthreads();

  // ---- R (only k<64 of A changes: kblks 0,1; H0 part kblks 2,3 untouched) ----
  for (int idx = tid; idx < 16 * 64; idx += 128) {
    int m = idx >> 6, k = idx & 63;
    As[afrag_idx(m, k)] = f2bf(gcnr[(n0 + m) * 64 + k] + br[k]);
  }
  __syncthreads();
  c = v8f{};
#pragma unroll
  for (int kb = 0; kb < 4; ++kb) {
    v16u a = load_afrag_swz_lds(As, lane, kb);
    v16u b = load_bfrag_swz(Lr, lane, kb, w, 4);
    c = wmma_bf16f32(a, b, c);
  }
  {
    float lb = lbr[col];
#pragma unroll
    for (int v = 0; v < 8; ++v) Rs[(rb + v) * 64 + col] = sigf(c[v] + lb);
  }
  __syncthreads();

  // ---- Ht ----
  for (int idx = tid; idx < 16 * 128; idx += 128) {
    int m = idx >> 7, k = idx & 127;
    float v = (k < 64) ? (gcnh[(n0 + m) * 64 + k] + bh[k])
                       : (H0s[m * 64 + (k - 64)] * Rs[m * 64 + (k - 64)]);
    As[afrag_idx(m, k)] = f2bf(v);
  }
  __syncthreads();
  c = v8f{};
#pragma unroll
  for (int kb = 0; kb < 4; ++kb) {
    v16u a = load_afrag_swz_lds(As, lane, kb);
    v16u b = load_bfrag_swz(Lh, lane, kb, w, 4);
    c = wmma_bf16f32(a, b, c);
  }
  {
    float lb = lbh[col];
#pragma unroll
    for (int v = 0; v < 8; ++v) {
      float ht = tanh_fast(c[v] + lb);
      float z  = Zs[(rb + v) * 64 + col];
      float h0 = H0s[(rb + v) * 64 + col];
      float hn = z * h0 + (1.0f - z) * ht;
      Hacc[(n0 + rb + v) * 64 + col] += prob * hn;   // exclusive ownership
    }
  }
}

// ---------------------------------------------------------------------------
// seq = kron(p,s) @ W_red^T + b_red   per 16-node tile, K=4096.
// A fragments built on the fly from p,s in LDS (kron never materialized);
// B fragments are single 32B loads from the swizzled W_red (L2 resident).
// ---------------------------------------------------------------------------
__global__ __launch_bounds__(256) void k_kron_gemm(
    const float* __restrict__ P, const float* __restrict__ S,
    const unsigned short* __restrict__ Wred,   // swizzled [4096][256]
    const float* __restrict__ bred,
    unsigned short* __restrict__ seq_swz) {    // per-tile 16x256 fragment order
  __shared__ float Ps[16 * 64];
  __shared__ float Ss[16 * 64];
  int tid = threadIdx.x, lane = tid & 31, w = tid >> 5;
  int n0 = blockIdx.x * 16;
  for (int idx = tid; idx < 16 * 64; idx += 256) {
    Ps[idx] = P[n0 * 64 + idx];
    Ss[idx] = S[n0 * 64 + idx];
  }
  __syncthreads();

  int m  = lane & 15;
  int kbh = (lane < 16) ? 0 : 8;
  int ct0 = w * 2, ct1 = w * 2 + 1;
  v8f c0 = {}, c1 = {};
  for (int kb = 0; kb < 128; ++kb) {
    int k0 = kb * 32;
    v16u a;
#pragma unroll
    for (int e = 0; e < 16; ++e) {
      int k = k0 + kbh + ((e < 8) ? e : (e + 8));
      a[e] = f2bf(Ps[m * 64 + (k >> 6)] * Ss[m * 64 + (k & 63)]);
    }
    if (kb + 4 < 128) {   // stream-ahead hint for W_red -> global_prefetch
      __builtin_prefetch(Wred + ((size_t)((kb + 4) * 16 + ct0) * 32 + lane) * 16, 0, 1);
    }
    v16u b0 = load_bfrag_swz(Wred, lane, kb, ct0, 16);
    v16u b1 = load_bfrag_swz(Wred, lane, kb, ct1, 16);
    c0 = wmma_bf16f32(a, b0, c0);
    c1 = wmma_bf16f32(a, b1, c1);
  }
  unsigned short* tile = seq_swz + (size_t)blockIdx.x * 16 * 256;
  int rbase = (lane < 16) ? 0 : 8;
  int col0 = ct0 * 16 + (lane & 15), col1 = ct1 * 16 + (lane & 15);
#pragma unroll
  for (int v = 0; v < 8; ++v) {
    tile[afrag_idx(rbase + v, col0)] = f2bf(c0[v] + bred[col0]);
    tile[afrag_idx(rbase + v, col1)] = f2bf(c1[v] + bred[col1]);
  }
}

// ---------------------------------------------------------------------------
// gates_x = seq @ W_ih^T   (N x 256) x (256 x 1024), fp32 out for the LSTM.
// ---------------------------------------------------------------------------
__global__ __launch_bounds__(256) void k_gatesx(
    const unsigned short* __restrict__ seq_swz,  // per-tile 16x256 fragments
    const unsigned short* __restrict__ Wih,      // swizzled [256][1024]
    float* __restrict__ gx) {                    // [N][1024]
  __shared__ __align__(32) unsigned short Asl[16 * 256];
  int tid = threadIdx.x, lane = tid & 31, w = tid >> 5;
  int n0 = blockIdx.x * 16;
  int cg = blockIdx.y;
  const uint4* src = (const uint4*)(seq_swz + (size_t)blockIdx.x * 16 * 256);
  uint4* dst = (uint4*)Asl;
  for (int idx = tid; idx < (16 * 256) / 8; idx += 256) dst[idx] = src[idx];
  __syncthreads();

  int ct0 = cg * 16 + w * 2, ct1 = ct0 + 1;   // CT = 64 col-tiles total
  v8f c0 = {}, c1 = {};
#pragma unroll
  for (int kb = 0; kb < 8; ++kb) {
    v16u a  = load_afrag_swz_lds(Asl, lane, kb);
    v16u b0 = load_bfrag_swz(Wih, lane, kb, ct0, 64);
    v16u b1 = load_bfrag_swz(Wih, lane, kb, ct1, 64);
    c0 = wmma_bf16f32(a, b0, c0);
    c1 = wmma_bf16f32(a, b1, c1);
  }
  int rbase = (lane < 16) ? 0 : 8;
  int col0 = ct0 * 16 + (lane & 15), col1 = ct1 * 16 + (lane & 15);
#pragma unroll
  for (int v = 0; v < 8; ++v) {
    gx[(size_t)(n0 + rbase + v) * 1024 + col0] = c0[v];
    gx[(size_t)(n0 + rbase + v) * 1024 + col1] = c1[v];
  }
}

// ---------------------------------------------------------------------------
// Sequential LSTM over N steps. One 1024-thread workgroup (32 waves / WGP).
// h,c,g live in LDS; each thread owns one gate row of Whh (bf16, L2 resident).
// Fast rcp-based nonlinearities shorten the serial dependence chain.
// ---------------------------------------------------------------------------
__global__ __launch_bounds__(1024) void k_lstm(
    const float* __restrict__ gx,             // [N][1024]
    const unsigned short* __restrict__ Whh,   // [1024][256] bf16 row-major
    const float* __restrict__ bih, const float* __restrict__ bhh,
    float* __restrict__ h_out, int nsteps) {
  __shared__ float h_lds[256];
  __shared__ float c_lds[256];
  __shared__ float g_lds[1024];
  int t = threadIdx.x;
  if (t < 256) { h_lds[t] = 0.0f; c_lds[t] = 0.0f; }
  __syncthreads();
  float bsum = bih[t] + bhh[t];
  const uint4* wrow4 = (const uint4*)(Whh + (size_t)t * 256);
  for (int s = 0; s < nsteps; ++s) {
    float acc = gx[(size_t)s * 1024 + t] + bsum;
#pragma unroll 4
    for (int k = 0; k < 256; k += 8) {
      uint4 u = wrow4[k >> 3];
      acc += h_lds[k + 0] * bf2f(u.x & 0xffffu) + h_lds[k + 1] * bf2f(u.x >> 16)
           + h_lds[k + 2] * bf2f(u.y & 0xffffu) + h_lds[k + 3] * bf2f(u.y >> 16)
           + h_lds[k + 4] * bf2f(u.z & 0xffffu) + h_lds[k + 5] * bf2f(u.z >> 16)
           + h_lds[k + 6] * bf2f(u.w & 0xffffu) + h_lds[k + 7] * bf2f(u.w >> 16);
    }
    g_lds[t] = acc;
    __syncthreads();
    if (t < 256) {
      float gi = g_lds[t], gf = g_lds[256 + t], gg = g_lds[512 + t], go = g_lds[768 + t];
      float cc = sigf(gf) * c_lds[t] + sigf(gi) * tanh_fast(gg);
      c_lds[t] = cc;
      h_lds[t] = sigf(go) * tanh_fast(cc);
    }
    __syncthreads();
  }
  if (t < 256) h_out[t] = h_lds[t];
}

// ---------------------------------------------------------------------------
// head: out = W_fc2 @ relu(W_fc1 @ h + b_fc1) + b_fc2
// ---------------------------------------------------------------------------
__global__ __launch_bounds__(256) void k_head(
    const float* __restrict__ h_last,
    const float* __restrict__ Wfc1, const float* __restrict__ bfc1,
    const float* __restrict__ Wfc2, const float* __restrict__ bfc2,
    float* __restrict__ out) {
  __shared__ float red[256];
  int t = threadIdx.x;
  float acc = bfc1[t];
  for (int k = 0; k < 256; ++k) acc += Wfc1[t * 256 + k] * h_last[k];
  float hid = fmaxf(acc, 0.0f);
  red[t] = Wfc2[t] * hid;
  __syncthreads();
  for (int s = 128; s > 0; s >>= 1) {
    if (t < s) red[t] += red[t + s];
    __syncthreads();
  }
  if (t == 0) out[0] = red[0] + bfc2[0];
}

// ---------------------------------------------------------------------------
// Host launch
// ---------------------------------------------------------------------------
extern "C" void kernel_launch(void* const* d_in, const int* in_sizes, int n_in,
                              void* d_out, int out_size, void* d_ws, size_t ws_size,
                              hipStream_t stream) {
  (void)n_in; (void)out_size; (void)ws_size;
  const int N = in_sizes[0] / (16 * 5);
  const int E = in_sizes[2] / 2;

  const float* x_price = (const float*)d_in[0];
  const float* x_sent  = (const float*)d_in[1];
  const int*   row     = (const int*)d_in[2];
  const int*   colp    = row + E;
  const float* eattr   = (const float*)d_in[3];
  const float* H0      = (const float*)d_in[4];
  // price_params: 5..17 ; sent_params: 18..30 ; head: 31..40
  const float* Wred  = (const float*)d_in[31];
  const float* bred  = (const float*)d_in[32];
  const float* Wih   = (const float*)d_in[33];
  const float* Whh   = (const float*)d_in[34];
  const float* bih   = (const float*)d_in[35];
  const float* bhh   = (const float*)d_in[36];
  const float* Wfc1  = (const float*)d_in[37];
  const float* bfc1  = (const float*)d_in[38];
  const float* Wfc2  = (const float*)d_in[39];
  const float* bfc2  = (const float*)d_in[40];

  // ---- workspace carve (256B aligned regions) ----
  char* wp = (char*)d_ws;
  auto carve = [&](size_t bytes) -> void* {
    void* r = (void*)wp;
    wp += (bytes + 255) & ~(size_t)255;
    return r;
  };
  float* deg      = (float*)carve((size_t)N * 4);
  float* dis      = (float*)carve((size_t)N * 4);
  float* selfnorm = (float*)carve((size_t)N * 4);
  float* normw    = (float*)carve((size_t)E * 4);
  float* xwz      = (float*)carve((size_t)N * 64 * 4);
  float* xwr      = (float*)carve((size_t)N * 64 * 4);
  float* xwh      = (float*)carve((size_t)N * 64 * 4);
  float* gz       = (float*)carve((size_t)N * 64 * 4);
  float* gr       = (float*)carve((size_t)N * 64 * 4);
  float* gh       = (float*)carve((size_t)N * 64 * 4);
  float* Hacc_p   = (float*)carve((size_t)N * 64 * 4);
  float* Hacc_s   = (float*)carve((size_t)N * 64 * 4);
  float* gatesx   = (float*)carve((size_t)N * 1024 * 4);
  float* h_last   = (float*)carve(256 * 4);
  float* probs    = (float*)carve(16 * 4);
  unsigned short* Lbf     = (unsigned short*)carve((size_t)6 * 128 * 64 * 2);
  unsigned short* Wred_bf = (unsigned short*)carve((size_t)4096 * 256 * 2);
  unsigned short* Wih_bf  = (unsigned short*)carve((size_t)256 * 1024 * 2);
  unsigned short* Whh_bf  = (unsigned short*)carve((size_t)1024 * 256 * 2);
  unsigned short* seq_swz = (unsigned short*)carve((size_t)N * 256 * 2);

  const int TB = 256;
  auto blocks = [&](long n) { return (unsigned)((n + TB - 1) / TB); };

  // ---- weight prep: bf16 + WMMA-fragment swizzles, attn softmax ----
  for (int s = 0; s < 2; ++s) {
    int base = (s == 0) ? 5 : 18;
    k_swizzleB<<<blocks(128 * 64), TB, 0, stream>>>((const float*)d_in[base + 3],
        Lbf + (size_t)(s * 3 + 0) * 128 * 64, 128, 64, 0);   // Lz [K][n]
    k_swizzleB<<<blocks(128 * 64), TB, 0, stream>>>((const float*)d_in[base + 7],
        Lbf + (size_t)(s * 3 + 1) * 128 * 64, 128, 64, 0);   // Lr
    k_swizzleB<<<blocks(128 * 64), TB, 0, stream>>>((const float*)d_in[base + 11],
        Lbf + (size_t)(s * 3 + 2) * 128 * 64, 128, 64, 0);   // Lh
  }
  k_swizzleB<<<blocks(4096L * 256), TB, 0, stream>>>(Wred, Wred_bf, 4096, 256, 1);
  k_swizzleB<<<blocks(256L * 1024), TB, 0, stream>>>(Wih, Wih_bf, 256, 1024, 1);
  k_cvt_bf16<<<blocks(1024L * 256), TB, 0, stream>>>(Whh, Whh_bf, 1024 * 256);
  k_softmax5<<<1, 32, 0, stream>>>((const float*)d_in[5], (const float*)d_in[18], probs);

  // ---- GCN normalization ----
  k_zero<<<blocks(N), TB, 0, stream>>>(deg, N);
  k_deg_edges<<<blocks(E), TB, 0, stream>>>(colp, eattr, deg, E);
  k_dis<<<blocks(N), TB, 0, stream>>>(deg, dis, selfnorm, N);
  k_edgenorm<<<blocks(E), TB, 0, stream>>>(row, colp, eattr, dis, normw, E);

  // ---- A3TGCN for both parameter sets ----
  k_zero<<<blocks((long)N * 64), TB, 0, stream>>>(Hacc_p, N * 64);
  k_zero<<<blocks((long)N * 64), TB, 0, stream>>>(Hacc_s, N * 64);
  int ntiles = (N + 15) / 16;
  for (int s = 0; s < 2; ++s) {
    const float* x = (s == 0) ? x_price : x_sent;
    int base = (s == 0) ? 5 : 18;
    const float* Wz  = (const float*)d_in[base + 1];
    const float* bz  = (const float*)d_in[base + 2];
    const float* lbz = (const float*)d_in[base + 4];
    const float* Wr  = (const float*)d_in[base + 5];
    const float* br  = (const float*)d_in[base + 6];
    const float* lbr = (const float*)d_in[base + 8];
    const float* Wh  = (const float*)d_in[base + 9];
    const float* bh  = (const float*)d_in[base + 10];
    const float* lbh = (const float*)d_in[base + 12];
    const unsigned short* Lz = Lbf + (size_t)(s * 3 + 0) * 128 * 64;
    const unsigned short* Lr = Lbf + (size_t)(s * 3 + 1) * 128 * 64;
    const unsigned short* Lh = Lbf + (size_t)(s * 3 + 2) * 128 * 64;
    float* Hacc = (s == 0) ? Hacc_p : Hacc_s;
    for (int t = 0; t < 5; ++t) {
      k_zero<<<blocks((long)N * 64), TB, 0, stream>>>(gz, N * 64);
      k_zero<<<blocks((long)N * 64), TB, 0, stream>>>(gr, N * 64);
      k_zero<<<blocks((long)N * 64), TB, 0, stream>>>(gh, N * 64);
      k_xw<<<blocks((long)N * 64), TB, 0, stream>>>(x, t, Wz, Wr, Wh, xwz, xwr, xwh, N);
      k_scatter<<<blocks((long)(E + N) * 16), TB, 0, stream>>>(
          row, colp, normw, selfnorm, xwz, xwr, xwh, gz, gr, gh, E, N);
      k_gate<<<ntiles, 128, 0, stream>>>(gz, gr, gh, bz, br, bh, H0,
                                         Lz, Lr, Lh, lbz, lbr, lbh,
                                         probs + s * 8, t, Hacc);
    }
  }

  // ---- kron + W_red reduction (WMMA, kron never materialized) ----
  k_kron_gemm<<<ntiles, 256, 0, stream>>>(Hacc_p, Hacc_s, Wred_bf, bred, seq_swz);

  // ---- LSTM input gates GEMM ----
  k_gatesx<<<dim3(ntiles, 4), 256, 0, stream>>>(seq_swz, Wih_bf, gatesx);

  // ---- sequential LSTM + head ----
  k_lstm<<<1, 1024, 0, stream>>>(gatesx, Whh_bf, bih, bhh, h_last, N);
  k_head<<<1, 256, 0, stream>>>(h_last, Wfc1, bfc1, Wfc2, bfc2, (float*)d_out);
}
